// NeuralODEMemory_28355374088720
// MI455X (gfx1250) — matrix operand order repaired and checked by reading, not compiled, not run on hardware
//
#include <hip/hip_runtime.h>
#include <hip/hip_bf16.h>

typedef __attribute__((ext_vector_type(16))) __bf16 v16bf;
typedef __attribute__((ext_vector_type(8)))  float  v8f;

#define DD        512
#define NROWS     64
#define HSTRIDE   516          // floats; 516 % 64 == 4 -> conflict-free column reads
#define USTRIDE   520          // bf16;   row pitch 1040B, 16B aligned, conflict-free
#define STEPS     10
#define DT_STEP   0.1f

union BFrag  { uint4 q[2]; v16bf v; };

__device__ __forceinline__ unsigned short f2bu(float f) {
    union { float f; unsigned u; } in; in.f = f;
    unsigned u = in.u;
    // round-to-nearest-even bf16
    return (unsigned short)((u + 0x7FFFu + ((u >> 16) & 1u)) >> 16);
}
__device__ __forceinline__ __bf16 f2bf(float f) {
    union { unsigned short s; __bf16 b; } o; o.s = f2bu(f); return o.b;
}
__device__ __forceinline__ float tanh_fast(float x) {
    float e = __expf(2.0f * x);
    return 1.0f - 2.0f / (e + 1.0f);
}
__device__ __forceinline__ float sigmoid_fast(float x) {
    return 1.0f / (1.0f + __expf(-x));
}

// ---------------------------------------------------------------------------
// Pack fp32 weight [512, kin] (row-major, torch Linear layout) into bf16
// B-fragment-contiguous layout: tile = ntile*kchunks + kc, each tile is
// 32 lanes x 16 bf16 (1KB). Lane l holds column n = ntile*16 + (l&15),
// elements k = kc*32 + (l>>4)*8 + (i&7) + (i>>3)*16  (i = 0..15). k >= kin -> 0.
// ---------------------------------------------------------------------------
__global__ void pack_weights(const float* __restrict__ W,
                             unsigned short* __restrict__ P,
                             int kin, int kchunks, int total) {
    int e = blockIdx.x * blockDim.x + threadIdx.x;
    if (e >= total) return;
    int i    = e & 15;
    int lane = (e >> 4) & 31;
    int tile = e >> 9;
    int ntile = tile / kchunks;
    int kc    = tile % kchunks;
    int n = ntile * 16 + (lane & 15);
    int k = kc * 32 + (lane >> 4) * 8 + (i & 7) + (i >> 3) * 16;
    float v = (k < kin) ? W[(size_t)n * kin + k] : 0.0f;
    P[e] = f2bu(v);
}

// ---------------------------------------------------------------------------
// Main fused kernel: 64-row tile per block, h resident in LDS for 10 Euler
// steps, all three GEMMs via v_wmma_f32_16x16x32_bf16.
// ---------------------------------------------------------------------------
__global__ __launch_bounds__(256)
void node_ode_kernel(const float* __restrict__ x,
                     const float* __restrict__ h0,
                     const float* __restrict__ b1,
                     const float* __restrict__ b2,
                     const float* __restrict__ bg,
                     const unsigned short* __restrict__ w1p,
                     const unsigned short* __restrict__ w2p,
                     const unsigned short* __restrict__ wgp,
                     float* __restrict__ out, int ntotal) {
    extern __shared__ char smem[];
    float*          hT = (float*)smem;
    unsigned short* uT = (unsigned short*)(smem + NROWS * HSTRIDE * 4);

    const int tid   = threadIdx.x;
    const int lane  = tid & 31;
    const int w     = tid >> 5;
    const int mband = (w & 3) * 16;        // 4 row-bands of 16
    const int nhalf = w >> 2;              // 2 column halves of 256
    const int lm    = lane & 15;
    const int kh    = lane >> 4;
    const int rowbase = blockIdx.x * NROWS;

    // ---- load h tile (fp32) into LDS, coalesced ----
    for (int i = tid; i < NROWS * DD; i += 256) {
        int r = i >> 9, c = i & 511;
        hT[r * HSTRIDE + c] = h0[(size_t)(rowbase + r) * DD + c];
    }
    __syncthreads();

    for (int s = 0; s < STEPS; ++s) {
        const float t = (float)s * DT_STEP;

        // ---- A fragments from h (fp32 LDS -> bf16 regs), rows mband..+15 ----
        v16bf af[17];
        {
            const float* hrow = &hT[(mband + lm) * HSTRIDE];
            #pragma unroll
            for (int kc = 0; kc < 16; ++kc) {
                int k0 = kc * 32 + kh * 8;
                v16bf a;
                #pragma unroll
                for (int i = 0; i < 8; ++i) {
                    a[i]     = f2bf(hrow[k0 + i]);
                    a[8 + i] = f2bf(hrow[k0 + 16 + i]);
                }
                af[kc] = a;
            }
            // K-chunk 16: time column at K=512 (a[0] on lanes 0-15), rest zero
            v16bf a;
            #pragma unroll
            for (int i = 0; i < 16; ++i) a[i] = f2bf(0.0f);
            a[0] = (kh == 0) ? f2bf(t) : f2bf(0.0f);
            af[16] = a;
        }

        // ---- layer 1: u = tanh(z @ W1^T + b1), u -> LDS bf16 ----
        #pragma unroll 1
        for (int nt = 0; nt < 16; ++nt) {
            int n0 = nhalf * 256 + nt * 16;
            v8f acc; 
            #pragma unroll
            for (int i = 0; i < 8; ++i) acc[i] = 0.0f;
            const char* bp = (const char*)w1p + ((size_t)(n0 >> 4) * 17) * 1024 + lane * 32;
            #pragma unroll
            for (int kc = 0; kc < 17; ++kc) {
                BFrag b;
                b.q[0] = *(const uint4*)(bp);
                b.q[1] = *(const uint4*)(bp + 16);
                bp += 1024;
                acc = __builtin_amdgcn_wmma_f32_16x16x32_bf16(
                          false, af[kc], false, b.v, (short)0, acc, false, false);
            }
            float bias = b1[n0 + lm];
            #pragma unroll
            for (int r = 0; r < 8; ++r) {
                int m = mband + kh * 8 + r;
                uT[m * USTRIDE + n0 + lm] = f2bu(tanh_fast(acc[r] + bias));
            }
        }
        __syncthreads();

        // ---- A fragments from u (bf16 LDS, direct b128 loads) ----
        v16bf uf[16];
        {
            const unsigned short* urow = &uT[(mband + lm) * USTRIDE];
            #pragma unroll
            for (int kc = 0; kc < 16; ++kc) {
                int k0 = kc * 32 + kh * 8;
                BFrag f;
                f.q[0] = *(const uint4*)(urow + k0);
                f.q[1] = *(const uint4*)(urow + k0 + 16);
                uf[kc] = f.v;
            }
        }

        // ---- layer 2: h += dt * tanh(u @ W2^T + b2) ----
        #pragma unroll 1
        for (int nt = 0; nt < 16; ++nt) {
            int n0 = nhalf * 256 + nt * 16;
            v8f acc;
            #pragma unroll
            for (int i = 0; i < 8; ++i) acc[i] = 0.0f;
            const char* bp = (const char*)w2p + ((size_t)(n0 >> 4) * 16) * 1024 + lane * 32;
            #pragma unroll
            for (int kc = 0; kc < 16; ++kc) {
                BFrag b;
                b.q[0] = *(const uint4*)(bp);
                b.q[1] = *(const uint4*)(bp + 16);
                bp += 1024;
                acc = __builtin_amdgcn_wmma_f32_16x16x32_bf16(
                          false, uf[kc], false, b.v, (short)0, acc, false, false);
            }
            float bias = b2[n0 + lm];
            #pragma unroll
            for (int r = 0; r < 8; ++r) {
                int m = mband + kh * 8 + r;
                hT[m * HSTRIDE + n0 + lm] += DT_STEP * tanh_fast(acc[r] + bias);
            }
        }
        __syncthreads();
    }

    // ---- gate: g = sigmoid([x, h] @ Wg^T + bg); out = g*h + (1-g)*x ----
    v16bf xf[16];
    {
        const float* xrow = &x[(size_t)(rowbase + mband + lm) * DD];
        #pragma unroll
        for (int kc = 0; kc < 16; ++kc) {
            int k0 = kc * 32 + kh * 8;
            v16bf a;
            #pragma unroll
            for (int i = 0; i < 8; ++i) {
                a[i]     = f2bf(xrow[k0 + i]);
                a[8 + i] = f2bf(xrow[k0 + 16 + i]);
            }
            xf[kc] = a;
        }
    }
    v16bf hf[16];
    {
        const float* hrow = &hT[(mband + lm) * HSTRIDE];
        #pragma unroll
        for (int kc = 0; kc < 16; ++kc) {
            int k0 = kc * 32 + kh * 8;
            v16bf a;
            #pragma unroll
            for (int i = 0; i < 8; ++i) {
                a[i]     = f2bf(hrow[k0 + i]);
                a[8 + i] = f2bf(hrow[k0 + 16 + i]);
            }
            hf[kc] = a;
        }
    }

    const size_t outoff = (size_t)ntotal * DD;   // second element of output tuple
    #pragma unroll 1
    for (int nt = 0; nt < 16; ++nt) {
        int n0 = nhalf * 256 + nt * 16;
        v8f acc;
        #pragma unroll
        for (int i = 0; i < 8; ++i) acc[i] = 0.0f;
        const char* bp = (const char*)wgp + ((size_t)(n0 >> 4) * 32) * 1024 + lane * 32;
        #pragma unroll
        for (int kc = 0; kc < 32; ++kc) {
            BFrag b;
            b.q[0] = *(const uint4*)(bp);
            b.q[1] = *(const uint4*)(bp + 16);
            bp += 1024;
            v16bf a = (kc < 16) ? xf[kc] : hf[kc - 16];
            acc = __builtin_amdgcn_wmma_f32_16x16x32_bf16(
                      false, a, false, b.v, (short)0, acc, false, false);
        }
        float bias = bg[n0 + lm];
        #pragma unroll
        for (int r = 0; r < 8; ++r) {
            int m   = mband + kh * 8 + r;
            int row = rowbase + m;
            int col = n0 + lm;
            float g  = sigmoid_fast(acc[r] + bias);
            float hv = hT[m * HSTRIDE + col];
            float xv = x[(size_t)row * DD + col];
            float o  = g * hv + (1.0f - g) * xv;
            size_t idx = (size_t)row * DD + col;
            out[idx]          = o;
            out[outoff + idx] = o;
        }
    }
}

extern "C" void kernel_launch(void* const* d_in, const int* in_sizes, int n_in,
                              void* d_out, int out_size, void* d_ws, size_t ws_size,
                              hipStream_t stream) {
    const float* x  = (const float*)d_in[0];
    const float* h0 = (const float*)d_in[1];
    const float* W1 = (const float*)d_in[2];
    const float* b1 = (const float*)d_in[3];
    const float* W2 = (const float*)d_in[4];
    const float* b2 = (const float*)d_in[5];
    const float* Wg = (const float*)d_in[6];
    const float* bg = (const float*)d_in[7];
    float* out = (float*)d_out;

    const int t1 = 32 * 17 * 512;   // W1 packed elems (K padded 513 -> 544)
    const int t2 = 32 * 16 * 512;   // W2 packed elems
    const int tg = 32 * 32 * 512;   // Wg packed elems
    unsigned short* w1p = (unsigned short*)d_ws;
    unsigned short* w2p = w1p + t1;
    unsigned short* wgp = w2p + t2;

    pack_weights<<<(t1 + 255) / 256, 256, 0, stream>>>(W1, w1p, 513, 17, t1);
    pack_weights<<<(t2 + 255) / 256, 256, 0, stream>>>(W2, w2p, 512, 16, t2);
    pack_weights<<<(tg + 255) / 256, 256, 0, stream>>>(Wg, wgp, 1024, 32, tg);

    const int nrows = in_sizes[0] / DD;                         // 65536
    const size_t lds = (size_t)NROWS * HSTRIDE * 4              // h fp32
                     + (size_t)NROWS * USTRIDE * 2;             // u bf16
    node_ode_kernel<<<nrows / NROWS, 256, lds, stream>>>(
        x, h0, b1, b2, bg, w1p, w2p, wgp, out, nrows);
}
